// SelfAttention_20143396618807
// MI455X (gfx1250) — matrix operand order repaired and checked by reading, not compiled, not run on hardware
//
#include <hip/hip_runtime.h>

using u16 = unsigned short;
using u32 = unsigned int;

typedef __attribute__((ext_vector_type(16))) __bf16 v16bf;
typedef __attribute__((ext_vector_type(8)))  float  v8f;

union Frag { u32 u[8]; v16bf v; };

__device__ __forceinline__ u16 f2bf(float f) {
  u32 u = __builtin_bit_cast(u32, f);
  u += 0x7FFFu + ((u >> 16) & 1u);   // round-to-nearest-even
  return (u16)(u >> 16);
}
__device__ __forceinline__ u32 pk2(float a, float b) {
  return (u32)f2bf(a) | ((u32)f2bf(b) << 16);
}
__device__ __forceinline__ v8f wmma_bf16(const Frag& a, const Frag& b, v8f c) {
  return __builtin_amdgcn_wmma_f32_16x16x32_bf16(false, a.v, false, b.v,
                                                 (short)0, c, false, false);
}

// ---------------------------------------------------------------------------
// GEMM: C[M,N] = A[M,K] * B[K,N] + bias   (A fp32 or bf16, C fp32 or bf16)
// 128x128x32 tiles, 256 threads (8 wave32), each wave computes 64x32.
// Software-pipelined: fetch tile t+1 to registers while computing tile t,
// ping-pong LDS buffers, one barrier per K-step.
// ---------------------------------------------------------------------------
template <typename AT, typename OT>
__global__ __launch_bounds__(256) void gemm_bias_kernel(
    const AT* __restrict__ A, const float* __restrict__ B,
    const float* __restrict__ bias, OT* __restrict__ C,
    int M, int N, int K)
{
  constexpr int BM = 128, BN = 128, BK = 32;
  __shared__ u16 As[2][BM][BK + 2];   // [buf][row][k]  bf16
  __shared__ u16 Bt[2][BN][BK + 2];   // [buf][col][k]  bf16 (transposed)

  const int tid  = threadIdx.x;
  const int lane = tid & 31;
  const int wid  = tid >> 5;
  const int hi   = lane >> 4;
  const int l16  = lane & 15;
  const int wm   = wid >> 2;
  const int wn   = wid & 3;
  const int mb   = blockIdx.x * BM;
  const int nb   = blockIdx.y * BN;

  uint2 aS[4], bS[4];   // staging registers (bf16 packed)

  auto fetch = [&](int kb) {
    #pragma unroll
    for (int i = 0; i < 4; i++) {
      int f = i * 256 + tid;
      int row = f >> 3, c4 = (f & 7) << 2;
      if constexpr (sizeof(AT) == 4) {
        const float4 a4 = *(const float4*)((const float*)A +
                            (size_t)(mb + row) * K + kb + c4);
        aS[i] = make_uint2(pk2(a4.x, a4.y), pk2(a4.z, a4.w));
      } else {
        aS[i] = *(const uint2*)((const u16*)A + (size_t)(mb + row) * K + kb + c4);
      }
      int kr = f >> 5, c4b = (f & 31) << 2;
      const float4 b4 = *(const float4*)(B + (size_t)(kb + kr) * N + nb + c4b);
      bS[i] = make_uint2(pk2(b4.x, b4.y), pk2(b4.z, b4.w));
    }
  };
  auto stage = [&](int buf) {
    #pragma unroll
    for (int i = 0; i < 4; i++) {
      int f = i * 256 + tid;
      int row = f >> 3, c4 = (f & 7) << 2;
      *(u32*)&As[buf][row][c4]     = aS[i].x;
      *(u32*)&As[buf][row][c4 + 2] = aS[i].y;
      int kr = f >> 5, c4b = (f & 31) << 2;
      Bt[buf][c4b + 0][kr] = (u16)(bS[i].x);
      Bt[buf][c4b + 1][kr] = (u16)(bS[i].x >> 16);
      Bt[buf][c4b + 2][kr] = (u16)(bS[i].y);
      Bt[buf][c4b + 3][kr] = (u16)(bS[i].y >> 16);
    }
  };

  v8f acc[4][2];
  #pragma unroll
  for (int i = 0; i < 4; i++)
    #pragma unroll
    for (int j = 0; j < 2; j++)
      #pragma unroll
      for (int r = 0; r < 8; r++) acc[i][j][r] = 0.f;

  fetch(0); stage(0);
  __syncthreads();

  const int nT = K / BK;
  for (int t = 0; t < nT; t++) {
    const int buf = t & 1;
    if (t + 1 < nT) fetch((t + 1) * BK);

    Frag af[4], bf[2];
    #pragma unroll
    for (int i = 0; i < 4; i++) {
      int row = wm * 64 + i * 16 + l16;
      #pragma unroll
      for (int j = 0; j < 8; j++) {
        int kk = 2 * j + (j >= 4 ? 8 : 0) + hi * 8;   // A-frag K interleave
        af[i].u[j] = *(const u32*)&As[buf][row][kk];
      }
    }
    #pragma unroll
    for (int j2 = 0; j2 < 2; j2++) {
      int col = wn * 32 + j2 * 16 + l16;
      #pragma unroll
      for (int j = 0; j < 8; j++) {
        int kk = 2 * j + (j >= 4 ? 8 : 0) + hi * 8;   // B-frag K layout
        bf[j2].u[j] = *(const u32*)&Bt[buf][col][kk];
      }
    }
    #pragma unroll
    for (int i = 0; i < 4; i++)
      #pragma unroll
      for (int j2 = 0; j2 < 2; j2++)
        acc[i][j2] = wmma_bf16(af[i], bf[j2], acc[i][j2]);

    if (t + 1 < nT) stage(buf ^ 1);
    __syncthreads();
  }

  #pragma unroll
  for (int j2 = 0; j2 < 2; j2++) {
    int col = nb + wn * 32 + j2 * 16 + l16;
    float bb = bias[col];
    #pragma unroll
    for (int i = 0; i < 4; i++) {
      #pragma unroll
      for (int r = 0; r < 8; r++) {
        int row = mb + wm * 64 + i * 16 + r + hi * 8;  // C-frag M mapping
        float val = acc[i][j2][r] + bb;
        if constexpr (sizeof(OT) == 4) C[(size_t)row * N + col] = val;
        else                           C[(size_t)row * N + col] = (OT)f2bf(val);
      }
    }
  }
}

// ---------------------------------------------------------------------------
// Flash attention: one block = (n, h, 128 queries); wave owns 16 queries.
// K tiles staged with GLOBAL_LOAD_ASYNC_TO_LDS_B128 (ASYNCcnt-tracked,
// XOR-swizzled LDS layout); V register-staged transposed. Double-buffered:
// tile t+1 in flight while tile t runs through WMMA + online softmax.
// ---------------------------------------------------------------------------
__global__ __launch_bounds__(256) void attn_kernel(
    const u16* __restrict__ q, const u16* __restrict__ k,
    const u16* __restrict__ v, u16* __restrict__ o, int S)
{
  constexpr int E = 1024, HD = 64;
  __shared__ alignas(16) u16 Ks[2][32 * HD];  // async-staged, swizzled row-major
  __shared__ u16 Vt[2][HD][32 + 2];           // [d][key] transposed
  __shared__ u16 Pw[8][16][32 + 2];           // per-wave P transpose scratch

  const int tid  = threadIdx.x;
  const int lane = tid & 31;
  const int wid  = tid >> 5;
  const int hi   = lane >> 4;
  const int l16  = lane & 15;
  const int n    = blockIdx.y >> 4;
  const int h    = blockIdx.y & 15;
  const int qb   = blockIdx.x * 128 + wid * 16;

  // Q fragments (16 queries x 64 dim = 2 A-frags), resident for whole loop
  Frag qf[2];
  #pragma unroll
  for (int c = 0; c < 2; c++)
    #pragma unroll
    for (int j = 0; j < 8; j++) {
      int kk = 2 * j + (j >= 4 ? 8 : 0) + hi * 8;
      qf[c].u[j] = *(const u32*)(q + (size_t)(n * S + qb + l16) * E +
                                 h * HD + c * 32 + kk);
    }

  // Async K staging addresses: 256 lanes x 16B = one 4KB tile per instruction.
  const int akey = tid >> 3;              // key row 0..31
  const int ac   = tid & 7;               // 16B chunk within row
  const u32 ksLds0 = (u32)(uintptr_t)(&Ks[0][0]) +
                     (u32)(akey * 128 + ((ac ^ (akey & 7)) << 4));  // swizzle
  const u32 kVoff  = (u32)(akey * (E * 2) + ac * 16);
  const unsigned long long kBase =
      (unsigned long long)(uintptr_t)k +
      ((unsigned long long)((size_t)(n * S) * E + h * HD)) * 2ull;

  auto issueK = [&](int kb, int buf) {
    unsigned long long sb = kBase + (unsigned long long)kb * (E * 2);
    u32 dst = ksLds0 + (u32)(buf * 32 * HD * 2);
    asm volatile("global_load_async_to_lds_b128 %0, %1, %2"
                 :: "v"(dst), "v"(kVoff), "s"(sb) : "memory");
  };
  auto loadVreg = [&](int kb, uint2* vr) {
    #pragma unroll
    for (int i = 0; i < 2; i++) {
      int f = i * 256 + tid;
      int key = f >> 4, d0 = (f & 15) << 2;
      vr[i] = *(const uint2*)(v + (size_t)(n * S + kb + key) * E + h * HD + d0);
    }
  };
  auto storeV = [&](int buf, const uint2* vr) {
    #pragma unroll
    for (int i = 0; i < 2; i++) {
      int f = i * 256 + tid;
      int key = f >> 4, d0 = (f & 15) << 2;
      Vt[buf][d0 + 0][key] = (u16)(vr[i].x);
      Vt[buf][d0 + 1][key] = (u16)(vr[i].x >> 16);
      Vt[buf][d0 + 2][key] = (u16)(vr[i].y);
      Vt[buf][d0 + 3][key] = (u16)(vr[i].y >> 16);
    }
  };

  float mrow[8], lrow[8];
  #pragma unroll
  for (int r = 0; r < 8; r++) { mrow[r] = -3.0e38f; lrow[r] = 0.f; }
  v8f oacc[4];
  #pragma unroll
  for (int c = 0; c < 4; c++)
    #pragma unroll
    for (int r = 0; r < 8; r++) oacc[c][r] = 0.f;

  {  // prologue: stage tile 0
    uint2 vr[2];
    issueK(0, 0);
    loadVreg(0, vr);
    storeV(0, vr);
    asm volatile("s_wait_asynccnt 0x0" ::: "memory");
  }
  __syncthreads();

  const int nT = S / 32;
  for (int t = 0; t < nT; t++) {
    const int buf = t & 1;
    uint2 vr[2];
    if (t + 1 < nT) {
      issueK((t + 1) * 32, buf ^ 1);     // async copy overlaps compute below
      loadVreg((t + 1) * 32, vr);
    }

    // S = Q * K^T : two 16x16 score tiles (keys 0-15, 16-31), K-dim 2x32
    v8f sf[2];
    #pragma unroll
    for (int s = 0; s < 2; s++)
      #pragma unroll
      for (int r = 0; r < 8; r++) sf[s][r] = 0.f;
    #pragma unroll
    for (int s = 0; s < 2; s++) {
      int key = s * 16 + l16;
      int sw  = key & 7;
      #pragma unroll
      for (int c = 0; c < 2; c++) {
        Frag kf;
        #pragma unroll
        for (int j = 0; j < 8; j++) {
          int d   = c * 32 + 2 * j + (j >= 4 ? 8 : 0) + hi * 8;
          int idx = key * HD + ((((d >> 3) ^ sw)) << 3) + (d & 7);
          kf.u[j] = *(const u32*)&Ks[buf][idx];
        }
        sf[s] = wmma_bf16(qf[c], kf, sf[s]);
      }
    }

    // online softmax (exp2 domain; scale = 1/sqrt(64) * log2(e))
    const float sc = 0.18033688f;
    #pragma unroll
    for (int r = 0; r < 8; r++) {
      float s0 = sf[0][r] * sc;
      float s1 = sf[1][r] * sc;
      float mx = fmaxf(s0, s1);
      #pragma unroll
      for (int off = 1; off < 16; off <<= 1)
        mx = fmaxf(mx, __shfl_xor(mx, off, 16));
      float mn = fmaxf(mrow[r], mx);
      float al = exp2f(mrow[r] - mn);
      mrow[r]  = mn;
      float p0 = exp2f(s0 - mn);
      float p1 = exp2f(s1 - mn);
      float ps = p0 + p1;
      #pragma unroll
      for (int off = 1; off < 16; off <<= 1)
        ps += __shfl_xor(ps, off, 16);
      lrow[r] = lrow[r] * al + ps;
      #pragma unroll
      for (int c2 = 0; c2 < 4; c2++) oacc[c2][r] *= al;
      int prow = r + hi * 8;
      Pw[wid][prow][l16]      = f2bf(p0);
      Pw[wid][prow][16 + l16] = f2bf(p1);
    }
    asm volatile("s_wait_dscnt 0" ::: "memory");  // C-layout -> A-layout via LDS

    Frag pf;
    #pragma unroll
    for (int j = 0; j < 8; j++) {
      int kk = 2 * j + (j >= 4 ? 8 : 0) + hi * 8;
      pf.u[j] = *(const u32*)&Pw[wid][l16][kk];
    }
    // O += P * V  (16x32 * 32x64, four 16-wide d-chunks)
    #pragma unroll
    for (int c = 0; c < 4; c++) {
      Frag vf;
      #pragma unroll
      for (int j = 0; j < 8; j++) {
        int kk = 2 * j + (j >= 4 ? 8 : 0) + hi * 8;
        vf.u[j] = *(const u32*)&Vt[buf][c * 16 + l16][kk];
      }
      oacc[c] = wmma_bf16(pf, vf, oacc[c]);
    }

    if (t + 1 < nT) storeV(buf ^ 1, vr);
    asm volatile("s_wait_asynccnt 0x0" ::: "memory");
    __syncthreads();
  }

  #pragma unroll
  for (int r = 0; r < 8; r++) {
    float rl = 1.0f / lrow[r];
    int row = qb + r + hi * 8;
    #pragma unroll
    for (int c = 0; c < 4; c++)
      o[(size_t)(n * S + row) * E + h * HD + c * 16 + l16] =
          f2bf(oacc[c][r] * rl);
  }
}

// ---------------------------------------------------------------------------
extern "C" void kernel_launch(void* const* d_in, const int* in_sizes, int n_in,
                              void* d_out, int out_size, void* d_ws, size_t ws_size,
                              hipStream_t stream) {
  (void)in_sizes; (void)n_in; (void)out_size; (void)ws_size;
  const float* values  = (const float*)d_in[0];
  const float* keys    = (const float*)d_in[1];
  const float* queries = (const float*)d_in[2];
  const float* Wv = (const float*)d_in[3];  const float* bv = (const float*)d_in[4];
  const float* Wk = (const float*)d_in[5];  const float* bk = (const float*)d_in[6];
  const float* Wq = (const float*)d_in[7];  const float* bq = (const float*)d_in[8];
  const float* Wo = (const float*)d_in[9];  const float* bo = (const float*)d_in[10];
  float* out = (float*)d_out;

  const int Nb = 2, S = 2048, E = 1024, H = 16;
  const int M = Nb * S;  // 4096

  u16* qws = (u16*)d_ws;
  u16* kws = qws + (size_t)M * E;
  u16* vws = kws + (size_t)M * E;
  u16* aws = vws + (size_t)M * E;

  dim3 blk(256);
  dim3 gg(M / 128, E / 128);
  gemm_bias_kernel<float, u16><<<gg, blk, 0, stream>>>(queries, Wq, bq, qws, M, E, E);
  gemm_bias_kernel<float, u16><<<gg, blk, 0, stream>>>(keys,    Wk, bk, kws, M, E, E);
  gemm_bias_kernel<float, u16><<<gg, blk, 0, stream>>>(values,  Wv, bv, vws, M, E, E);

  dim3 ga(S / 128, Nb * H);
  attn_kernel<<<ga, blk, 0, stream>>>(qws, kws, vws, aws, S);

  gemm_bias_kernel<u16, float><<<gg, blk, 0, stream>>>(aws, Wo, bo, out, M, E, E);
}
